// Encoder_66640712565349
// MI455X (gfx1250) — matrix operand order repaired and checked by reading, not compiled, not run on hardware
//
#include <hip/hip_runtime.h>
#include <hip/hip_bf16.h>

// ---------------------------------------------------------------------------
// Problem constants (from reference): B=8, L=4096, D=1024, H=16, DH=64, FF=4
// ---------------------------------------------------------------------------
#define B_   8
#define L_   4096
#define D_   1024
#define H_   16
#define DH_  64
#define FFD_ 4096
#define M_   (B_ * L_)        // 32768 rows

typedef __attribute__((ext_vector_type(16))) __bf16 v16bf;
typedef __attribute__((ext_vector_type(8)))  __bf16 v8bf;
typedef __attribute__((ext_vector_type(8)))  float  v8f;

// ---------------------------------------------------------------------------
// Weight convert + transpose: W (K x N, f32 row-major) -> Wt (N x K, bf16)
// ---------------------------------------------------------------------------
__global__ __launch_bounds__(256)
void wt_to_bf16_t(const float* __restrict__ W, __bf16* __restrict__ Wt,
                  int Kd, int Nd)
{
    size_t i = (size_t)blockIdx.x * 256 + threadIdx.x;   // over Nd*Kd
    int n = (int)(i / Kd);
    int k = (int)(i % Kd);
    Wt[i] = (__bf16)W[(size_t)k * Nd + n];
}

// ---------------------------------------------------------------------------
// X = X + pe ; keep fp32 residual and bf16 GEMM operand
// ---------------------------------------------------------------------------
__global__ __launch_bounds__(256)
void add_pe(const float* __restrict__ X, const float* __restrict__ pe,
            float* __restrict__ Xres, __bf16* __restrict__ Xbf)
{
    size_t i = (size_t)blockIdx.x * 256 + threadIdx.x;   // over M_*D_
    float v = X[i] + pe[i % ((size_t)L_ * D_)];
    Xres[i] = v;
    Xbf[i]  = (__bf16)v;
}

// ---------------------------------------------------------------------------
// Fragment loaders (per CDNA5 ISA 16-bit WMMA VGPR layouts, wave32)
// A 16x32: lanes 0-15 hold K 0..7 / 16..23 ; lanes 16-31 hold K 8..15 / 24..31
// B 32x16 (from N x K transposed weights): lanes 0-15 K 0..7 / 16..23 ; lanes 16-31 K 8..15 / 24..31
// ---------------------------------------------------------------------------
__device__ __forceinline__ v16bf load_frag_a(const __bf16* p)
{
    v8bf lo = *(const v8bf*)(p);
    v8bf hi = *(const v8bf*)(p + 16);
    v16bf r;
#pragma unroll
    for (int i = 0; i < 8; ++i) { r[i] = lo[i]; r[8 + i] = hi[i]; }
    return r;
}

__device__ __forceinline__ v16bf load_frag_b(const __bf16* p)
{
    v8bf lo = *(const v8bf*)(p);
    v8bf hi = *(const v8bf*)(p + 16);
    v16bf r;
#pragma unroll
    for (int i = 0; i < 8; ++i) { r[i] = lo[i]; r[8 + i] = hi[i]; }
    return r;
}

// ---------------------------------------------------------------------------
// WMMA GEMM:  C[M,N] = A[M,K](bf16) * Bt[N,K](bf16)^T + bias
// Block: 256 threads = 8 waves (2 M x 4 N); block tile 128x256; wave tile
// 64x64 (4x4 WMMA tiles -> 32 FLOP/byte from cache in the steady state).
// Compile-time NDIM/KDIM/MODE: MODE 0 -> fp32 out; MODE 1 -> relu + bf16 out.
// K-loop double-buffered (unroll 64) so loads overlap WMMA issue.
// ---------------------------------------------------------------------------
template <int NDIM, int KDIM, int MODE>
__global__ __launch_bounds__(256)
void gemm_bf16_wmma(const __bf16* __restrict__ A,
                    const __bf16* __restrict__ Bt,
                    const float*  __restrict__ bias,
                    float*        __restrict__ Cf,
                    __bf16*       __restrict__ Cbf)
{
    const int lane = threadIdx.x & 31;
    const int wid  = threadIdx.x >> 5;      // 0..7
    const int hlf  = lane >> 4;             // 0/1
    const int l16  = lane & 15;
    const int bm   = blockIdx.y * 128 + (wid >> 2) * 64;   // wave row base
    const int bn   = blockIdx.x * 256 + (wid & 3) * 64;    // wave col base

    const __bf16* ap[4];
    const __bf16* bp[4];
#pragma unroll
    for (int mt = 0; mt < 4; ++mt)
        ap[mt] = A + (size_t)(bm + mt * 16 + l16) * KDIM + hlf * 8;
#pragma unroll
    for (int nt = 0; nt < 4; ++nt)
        bp[nt] = Bt + (size_t)(bn + nt * 16 + l16) * KDIM + hlf * 8;

    v8f c[4][4];
#pragma unroll
    for (int mt = 0; mt < 4; ++mt)
#pragma unroll
        for (int nt = 0; nt < 4; ++nt) {
            v8f z = {0.f, 0.f, 0.f, 0.f, 0.f, 0.f, 0.f, 0.f};
            c[mt][nt] = z;
        }

    v16bf a0[4], a1[4], b0[4], b1[4];

    // Prologue: load K-slab 0
#pragma unroll
    for (int mt = 0; mt < 4; ++mt) a0[mt] = load_frag_a(ap[mt]);
#pragma unroll
    for (int nt = 0; nt < 4; ++nt) b0[nt] = load_frag_b(bp[nt]);

#define WMMA_STEP(AF, BF)                                                     \
    _Pragma("unroll")                                                         \
    for (int mt = 0; mt < 4; ++mt)                                            \
        _Pragma("unroll")                                                     \
        for (int nt = 0; nt < 4; ++nt)                                        \
            c[mt][nt] = __builtin_amdgcn_wmma_f32_16x16x32_bf16(              \
                false, AF[mt], false, BF[nt], (short)0, c[mt][nt],            \
                false, false);

    // Steady state: two 32-wide K slabs per iteration, ping-pong buffers.
    for (int kb = 0; kb + 64 < KDIM; kb += 64) {
#pragma unroll
        for (int mt = 0; mt < 4; ++mt) {
            a1[mt] = load_frag_a(ap[mt] + kb + 32);
            __builtin_prefetch(ap[mt] + kb + 64, 0, 3);
        }
#pragma unroll
        for (int nt = 0; nt < 4; ++nt) {
            b1[nt] = load_frag_b(bp[nt] + kb + 32);
            __builtin_prefetch(bp[nt] + kb + 64, 0, 3);
        }
        WMMA_STEP(a0, b0)
#pragma unroll
        for (int mt = 0; mt < 4; ++mt) a0[mt] = load_frag_a(ap[mt] + kb + 64);
#pragma unroll
        for (int nt = 0; nt < 4; ++nt) b0[nt] = load_frag_b(bp[nt] + kb + 64);
        WMMA_STEP(a1, b1)
    }

    // Epilogue of K loop: final 64 (KDIM % 64 == 0, KDIM >= 64)
#pragma unroll
    for (int mt = 0; mt < 4; ++mt) a1[mt] = load_frag_a(ap[mt] + KDIM - 32);
#pragma unroll
    for (int nt = 0; nt < 4; ++nt) b1[nt] = load_frag_b(bp[nt] + KDIM - 32);
    WMMA_STEP(a0, b0)
    WMMA_STEP(a1, b1)
#undef WMMA_STEP

    // Store. C/D f32 layout: VGPR r -> row (r + hlf*8), col l16.
#pragma unroll
    for (int mt = 0; mt < 4; ++mt)
#pragma unroll
        for (int nt = 0; nt < 4; ++nt) {
            const int col = bn + nt * 16 + l16;
            const int row = bm + mt * 16 + hlf * 8;
            const float bval = bias[col];
            if (MODE == 0) {
                float* out = Cf + (size_t)row * NDIM + col;
#pragma unroll
                for (int r = 0; r < 8; ++r)
                    out[(size_t)r * NDIM] = c[mt][nt][r] + bval;
            } else {
                __bf16* out = Cbf + (size_t)row * NDIM + col;
#pragma unroll
                for (int r = 0; r < 8; ++r)
                    out[(size_t)r * NDIM] =
                        (__bf16)fmaxf(c[mt][nt][r] + bval, 0.f);
            }
        }
}

// ---------------------------------------------------------------------------
// Window-3 local attention. One wave per (b, l, h); 2 channels per lane.
// ---------------------------------------------------------------------------
__global__ __launch_bounds__(256)
void attn_win3(const float* __restrict__ Q, const float* __restrict__ Km,
               const float* __restrict__ V, const unsigned char* __restrict__ mask,
               __bf16* __restrict__ ctx)
{
    const int wid  = blockIdx.x * 8 + (threadIdx.x >> 5);  // over B*L*H
    const int lane = threadIdx.x & 31;
    const int h = wid % H_;
    const int t = wid / H_;
    const int l = t % L_;
    const int b = t / L_;

    const size_t row  = (size_t)b * L_ + l;
    const size_t coff = (size_t)h * DH_ + lane * 2;
    const float* qp = Q + row * D_ + coff;
    const float qx = qp[0], qy = qp[1];

    float sc[3], vx[3], vy[3];
#pragma unroll
    for (int w = 0; w < 3; ++w) {
        const int idx  = l + w - 1;
        const int idxc = min(max(idx, 0), L_ - 1);
        const bool ok  = (idx >= 0) && (idx < L_) &&
                         (mask[(size_t)b * L_ + idxc] == 0);
        const size_t rc = (size_t)b * L_ + idxc;
        const float* kp = Km + rc * D_ + coff;
        const float* vp = V  + rc * D_ + coff;
        float p = qx * kp[0] + qy * kp[1];
        vx[w] = vp[0];
        vy[w] = vp[1];
#pragma unroll
        for (int off = 16; off; off >>= 1) p += __shfl_xor(p, off, 32);
        sc[w] = ok ? p * 0.125f : -1e30f;    // 1/sqrt(DH=64) = 0.125
    }

    const float m  = fmaxf(sc[0], fmaxf(sc[1], sc[2]));
    const float e0 = __expf(sc[0] - m);
    const float e1 = __expf(sc[1] - m);
    const float e2 = __expf(sc[2] - m);
    const float inv = 1.f / (e0 + e1 + e2);
    const float cx = (e0 * vx[0] + e1 * vx[1] + e2 * vx[2]) * inv;
    const float cy = (e0 * vy[0] + e1 * vy[1] + e2 * vy[2]) * inv;

    __bf16* op = ctx + row * D_ + coff;
    op[0] = (__bf16)cx;
    op[1] = (__bf16)cy;
}

// ---------------------------------------------------------------------------
// y = LayerNorm(res + x) * g + beta ; block per row, 4 cols/thread.
// ---------------------------------------------------------------------------
__global__ __launch_bounds__(256)
void ln_residual(const float* __restrict__ res, const float* __restrict__ x,
                 const float* __restrict__ g, const float* __restrict__ beta,
                 float* __restrict__ outf, __bf16* __restrict__ outbf)
{
    __shared__ float red[256];
    const int tid = threadIdx.x;
    const size_t row = blockIdx.x;
    const float* rp = res + row * D_;
    const float* xp = x   + row * D_;

    float v[4];
    float s = 0.f;
#pragma unroll
    for (int i = 0; i < 4; ++i) {
        const int col = tid + i * 256;
        v[i] = rp[col] + xp[col];
        s += v[i];
    }
    red[tid] = s;
    __syncthreads();
    for (int off = 128; off; off >>= 1) {
        if (tid < off) red[tid] += red[tid + off];
        __syncthreads();
    }
    const float mean = red[0] * (1.0f / D_);
    __syncthreads();

    s = 0.f;
#pragma unroll
    for (int i = 0; i < 4; ++i) { const float d = v[i] - mean; s += d * d; }
    red[tid] = s;
    __syncthreads();
    for (int off = 128; off; off >>= 1) {
        if (tid < off) red[tid] += red[tid + off];
        __syncthreads();
    }
    const float rstd = rsqrtf(red[0] * (1.0f / D_) + 1e-5f);

    float y[4];
#pragma unroll
    for (int i = 0; i < 4; ++i) {
        const int col = tid + i * 256;
        y[i] = (v[i] - mean) * rstd * g[col] + beta[col];
        outf[row * D_ + col] = y[i];
    }
    if (outbf != nullptr) {
#pragma unroll
        for (int i = 0; i < 4; ++i)
            outbf[row * D_ + tid + i * 256] = (__bf16)y[i];
    }
}

// ---------------------------------------------------------------------------
// Host-side orchestration
// ---------------------------------------------------------------------------
extern "C" void kernel_launch(void* const* d_in, const int* in_sizes, int n_in,
                              void* d_out, int out_size, void* d_ws, size_t ws_size,
                              hipStream_t stream)
{
    (void)in_sizes; (void)n_in; (void)out_size; (void)ws_size;

    const float* X    = (const float*)d_in[0];
    const unsigned char* mask = (const unsigned char*)d_in[1];
    const float* pe   = (const float*)d_in[2];
    const float* Wq   = (const float*)d_in[3];
    const float* Wk   = (const float*)d_in[4];
    const float* Wv   = (const float*)d_in[5];
    const float* bq   = (const float*)d_in[6];
    const float* bk   = (const float*)d_in[7];
    const float* bv   = (const float*)d_in[8];
    const float* Wo   = (const float*)d_in[9];
    const float* bo   = (const float*)d_in[10];
    const float* g1   = (const float*)d_in[11];
    const float* be1  = (const float*)d_in[12];
    const float* W1   = (const float*)d_in[13];
    const float* b1   = (const float*)d_in[14];
    const float* W2   = (const float*)d_in[15];
    const float* b2   = (const float*)d_in[16];
    const float* g2   = (const float*)d_in[17];
    const float* be2  = (const float*)d_in[18];

    char* ws = (char*)d_ws;
    const size_t MB = (size_t)1 << 20;

    // Workspace layout (lifetime-overlapped; peak = 856 MB)
    __bf16* WQt   = (__bf16*)(ws + 0 * MB);     //  2 MB
    __bf16* WKt   = (__bf16*)(ws + 2 * MB);     //  2 MB
    __bf16* WVt   = (__bf16*)(ws + 4 * MB);     //  2 MB
    __bf16* WOt   = (__bf16*)(ws + 6 * MB);     //  2 MB
    __bf16* W1t   = (__bf16*)(ws + 8 * MB);     //  8 MB
    __bf16* W2t   = (__bf16*)(ws + 16 * MB);    //  8 MB
    float*  Xres  = (float*) (ws + 24 * MB);    // 128 MB
    __bf16* Xbf   = (__bf16*)(ws + 152 * MB);   //  64 MB
    float*  Qb    = (float*) (ws + 216 * MB);   // 128 MB
    float*  Kb    = (float*) (ws + 344 * MB);   // 128 MB
    float*  Vb    = (float*) (ws + 472 * MB);   // 128 MB
    __bf16* ctxbf = (__bf16*)(ws + 152 * MB);   // reuse Xbf
    float*  Tb    = (float*) (ws + 216 * MB);   // reuse Qb
    float*  X1res = (float*) (ws + 344 * MB);   // reuse Kb
    __bf16* X1bf  = (__bf16*)(ws + 472 * MB);   // reuse Vb
    __bf16* Fbf   = (__bf16*)(ws + 600 * MB);   // 256 MB
    float*  F2b   = (float*) (ws + 216 * MB);   // reuse Tb

    // 1) Weight conversion / transpose (tiny: ~12M elements total)
    wt_to_bf16_t<<<(D_ * D_) / 256, 256, 0, stream>>>(Wq, WQt, D_, D_);
    wt_to_bf16_t<<<(D_ * D_) / 256, 256, 0, stream>>>(Wk, WKt, D_, D_);
    wt_to_bf16_t<<<(D_ * D_) / 256, 256, 0, stream>>>(Wv, WVt, D_, D_);
    wt_to_bf16_t<<<(D_ * D_) / 256, 256, 0, stream>>>(Wo, WOt, D_, D_);
    wt_to_bf16_t<<<(D_ * FFD_) / 256, 256, 0, stream>>>(W1, W1t, D_, FFD_);
    wt_to_bf16_t<<<(FFD_ * D_) / 256, 256, 0, stream>>>(W2, W2t, FFD_, D_);

    // 2) X = X + pe
    add_pe<<<((size_t)M_ * D_) / 256, 256, 0, stream>>>(X, pe, Xres, Xbf);

    // 3) QKV projections (WMMA bf16, f32 accumulate)
    dim3 gD(D_ / 256, M_ / 128);       // (4, 256)
    dim3 gF(FFD_ / 256, M_ / 128);     // (16, 256)
    gemm_bf16_wmma<D_, D_, 0><<<gD, 256, 0, stream>>>(Xbf, WQt, bq, Qb, nullptr);
    gemm_bf16_wmma<D_, D_, 0><<<gD, 256, 0, stream>>>(Xbf, WKt, bk, Kb, nullptr);
    gemm_bf16_wmma<D_, D_, 0><<<gD, 256, 0, stream>>>(Xbf, WVt, bv, Vb, nullptr);

    // 4) Window-3 attention -> bf16 ctx
    attn_win3<<<(M_ * H_) / 8, 256, 0, stream>>>(Qb, Kb, Vb, mask, ctxbf);

    // 5) Output projection, then residual + LayerNorm 1
    gemm_bf16_wmma<D_, D_, 0><<<gD, 256, 0, stream>>>(ctxbf, WOt, bo, Tb, nullptr);
    ln_residual<<<M_, 256, 0, stream>>>(Xres, Tb, g1, be1, X1res, X1bf);

    // 6) FFN: GEMM1 fused ReLU->bf16, GEMM2, residual + LayerNorm 2 -> d_out
    gemm_bf16_wmma<FFD_, D_, 1><<<gF, 256, 0, stream>>>(X1bf, W1t, b1, nullptr, Fbf);
    gemm_bf16_wmma<D_, FFD_, 0><<<gD, 256, 0, stream>>>(Fbf, W2t, b2, F2b, nullptr);
    ln_residual<<<M_, 256, 0, stream>>>(X1res, F2b, g2, be2, (float*)d_out, nullptr);
}
